// SpikingSelfAttention_24309514895933
// MI455X (gfx1250) — matrix operand order repaired and checked by reading, not compiled, not run on hardware
//
#include <hip/hip_runtime.h>

// ---------------------------------------------------------------------------
// Spiking self-attention (T=4, B=16, N=1024, D=512, H=8) for gfx1250 (MI455X)
// f16 WMMA pipeline, async global->LDS double-buffered A-operand staging,
// fully-fused LIF + causal linear attention epilogue.
// ---------------------------------------------------------------------------

typedef _Float16 h8   __attribute__((ext_vector_type(8)));
typedef _Float16 v16h __attribute__((ext_vector_type(16)));
typedef float    v8f  __attribute__((ext_vector_type(8)));
typedef float    f4   __attribute__((ext_vector_type(4)));

#define T_STEPS 4
#define DIM     512
#define MROWS   16384            // B*N
#define TOTROWS (T_STEPS * MROWS)
#define BK      32               // K-chunk per WMMA
#define NKC     (DIM / BK)       // 16 k-chunks

#define LIF_DECAY  0.9f
#define LIF_THRESH 1.0f

#define WMMA_F16(a, b, c) \
  __builtin_amdgcn_wmma_f32_16x16x32_f16(false, (a), false, (b), (short)0, (c), false, false)

// ---- CDNA5 async global->LDS copy (ASYNCcnt-tracked), inline asm ----------
__device__ __forceinline__ void async_ld16(unsigned lds_off, const _Float16* g) {
  asm volatile("global_load_async_to_lds_b128 %0, %1, off"
               :: "v"(lds_off), "v"(g)
               : "memory");
}
__device__ __forceinline__ void wait_async_le4() {
  asm volatile("s_wait_asynccnt 0x4" ::: "memory");
}
__device__ __forceinline__ void wait_async_0() {
  asm volatile("s_wait_asynccnt 0x0" ::: "memory");
}

// A-fragment (16x32, M x K) from a row-major f16 tile with row stride BK.
// Lanes 0-15: row mloc+lane,  K = {0..7, 16..23} of the chunk
// Lanes 16-31: row mloc+lane-16, K = {8..15, 24..31}
__device__ __forceinline__ v16h load_frag_a_lds(const _Float16* tile,
                                                int mloc, int lhalf) {
  const _Float16* p = tile + mloc * BK + lhalf * 8;
  h8 lo = *(const h8*)(p);
  h8 hi = *(const h8*)(p + 16);
  return __builtin_shufflevector(lo, hi, 0, 1, 2, 3, 4, 5, 6, 7,
                                 8, 9, 10, 11, 12, 13, 14, 15);
}

// A-fragment directly from global row-major f16 (row stride DIM).
__device__ __forceinline__ v16h load_frag_a(const _Float16* __restrict__ base,
                                            int row, int k0, int lhalf) {
  const _Float16* p = base + (size_t)row * DIM + k0 + lhalf * 8;
  h8 lo = *(const h8*)(p);
  h8 hi = *(const h8*)(p + 16);
  return __builtin_shufflevector(lo, hi, 0, 1, 2, 3, 4, 5, 6, 7,
                                 8, 9, 10, 11, 12, 13, 14, 15);
}

// B-fragment (32x16, K x N) built from row-major W (row n holds K contiguous).
// Lanes 0-15: col n0+lane, K = {0..7, 16..23} ; lanes 16-31: K = {8..15, 24..31}
__device__ __forceinline__ v16h load_frag_b(const _Float16* __restrict__ base,
                                            int row, int k0, int lhalf) {
  const _Float16* p = base + (size_t)row * DIM + k0 + lhalf * 8;
  h8 lo = *(const h8*)(p);
  h8 hi = *(const h8*)(p + 16);
  return __builtin_shufflevector(lo, hi, 0, 1, 2, 3, 4, 5, 6, 7,
                                 8, 9, 10, 11, 12, 13, 14, 15);
}

// --------------------------- conversion kernel -----------------------------
__global__ void cvt_f32_to_f16_kernel(const float* __restrict__ src,
                                      _Float16* __restrict__ dst, int n8) {
  int i = blockIdx.x * blockDim.x + threadIdx.x;
  if (i >= n8) return;
  const f4* s = (const f4*)src + (size_t)i * 2;
  f4 a = s[0];
  f4 b = s[1];
  h8 r = {(_Float16)a[0], (_Float16)a[1], (_Float16)a[2], (_Float16)a[3],
          (_Float16)b[0], (_Float16)b[1], (_Float16)b[2], (_Float16)b[3]};
  *((h8*)dst + i) = r;
}

// ------------------- fused q/k/v GEMM + LIF + attention --------------------
// Block: 256 threads = 8 waves; wave grid 4(M) x 2(N); block tile 64x32.
// x tiles (all 4 timesteps) staged in LDS via async copies, double-buffered.
// Each wave: one 16x16 output tile, accumulators for all T and all 3 projs.
__global__ __launch_bounds__(256) void qkv_lif_attn_kernel(
    const _Float16* __restrict__ xh,   // (T, MROWS, DIM) f16
    const _Float16* __restrict__ wq,   // (DIM, DIM) f16 row-major
    const _Float16* __restrict__ wk,
    const _Float16* __restrict__ wv,
    _Float16* __restrict__ ah) {       // (T, MROWS, DIM) f16 : q .* cumsum(k.*v)
  __shared__ __align__(16) _Float16 tileA[2][T_STEPS][64 * BK];  // 32 KB

  const int tid   = threadIdx.x;
  const int lane  = tid & 31;
  const int wave  = tid >> 5;
  const int wm    = wave & 3;
  const int wn    = wave >> 2;
  const int row0  = blockIdx.x * 64;       // block row base
  const int m0    = row0 + wm * 16;
  const int n0    = blockIdx.y * 32 + wn * 16;
  const int lrow  = lane & 15;
  const int lhalf = lane >> 4;
  const int mloc  = wm * 16 + lrow;        // A row within the LDS tile

  // Per-thread async-copy assignment: one 16B chunk per timestep per k-chunk.
  const int crow  = tid >> 2;              // 0..63 tile row
  const int ckseg = (tid & 3) * 8;         // f16 offset within chunk

  v8f accQ[T_STEPS], accK[T_STEPS], accV[T_STEPS];
#pragma unroll
  for (int t = 0; t < T_STEPS; ++t) {
    accQ[t] = (v8f){0.f, 0.f, 0.f, 0.f, 0.f, 0.f, 0.f, 0.f};
    accK[t] = accQ[t];
    accV[t] = accQ[t];
  }

  const int brow = n0 + lrow;              // B-fragment row = output column

  // Issue async copies of chunk 0 into buffer 0.
#pragma unroll
  for (int t = 0; t < T_STEPS; ++t) {
    const _Float16* g =
        xh + (size_t)t * MROWS * DIM + (size_t)(row0 + crow) * DIM + ckseg;
    async_ld16((unsigned)(uintptr_t)&tileA[0][t][crow * BK + ckseg], g);
  }

  for (int kc = 0; kc < NKC; ++kc) {
    const int k0 = kc * BK;
    if (kc + 1 < NKC) {
      // Prefetch next chunk into the other buffer (its readers finished at
      // the barrier that ended iteration kc-1).
      const int nb = (kc + 1) & 1;
      const int nk0 = k0 + BK;
#pragma unroll
      for (int t = 0; t < T_STEPS; ++t) {
        const _Float16* g =
            xh + (size_t)t * MROWS * DIM + (size_t)(row0 + crow) * DIM + nk0 + ckseg;
        async_ld16((unsigned)(uintptr_t)&tileA[nb][t][crow * BK + ckseg], g);
      }
      wait_async_le4();   // batch kc has landed (ASYNCcnt completes in order)
    } else {
      wait_async_0();
    }
    __syncthreads();      // make this chunk visible to all 8 waves

    v16h bq = load_frag_b(wq, brow, k0, lhalf);
    v16h bk = load_frag_b(wk, brow, k0, lhalf);
    v16h bv = load_frag_b(wv, brow, k0, lhalf);
#pragma unroll
    for (int t = 0; t < T_STEPS; ++t) {
      v16h at = load_frag_a_lds(&tileA[kc & 1][t][0], mloc, lhalf);
      accQ[t] = WMMA_F16(at, bq, accQ[t]);
      accK[t] = WMMA_F16(at, bk, accK[t]);
      accV[t] = WMMA_F16(at, bv, accV[t]);
    }
    __syncthreads();      // all waves done reading buf[kc&1] before rewrite
  }

  // LIF recurrence (per element) + k.*v cumsum + q.*context, all in registers.
  // C/D layout: VGPR r of lanes 0-15 -> (M = m0+r,    N = n0+lane)
  //             VGPR r of lanes 16-31 -> (M = m0+8+r, N = n0+lane-16)
  v8f memq = (v8f){0.f, 0.f, 0.f, 0.f, 0.f, 0.f, 0.f, 0.f};
  v8f memk = memq, memv = memq, ctx = memq;

#pragma unroll
  for (int t = 0; t < T_STEPS; ++t) {
    _Float16* arow_base = ah + (size_t)t * MROWS * DIM;
#pragma unroll
    for (int i = 0; i < 8; ++i) {
      memq[i] = LIF_DECAY * memq[i] + accQ[t][i];
      float sq = (memq[i] >= LIF_THRESH) ? 1.0f : 0.0f;
      memq[i] -= sq * LIF_THRESH;

      memk[i] = LIF_DECAY * memk[i] + accK[t][i];
      float sk = (memk[i] >= LIF_THRESH) ? 1.0f : 0.0f;
      memk[i] -= sk * LIF_THRESH;

      memv[i] = LIF_DECAY * memv[i] + accV[t][i];
      float sv = (memv[i] >= LIF_THRESH) ? 1.0f : 0.0f;
      memv[i] -= sv * LIF_THRESH;

      ctx[i] += sk * sv;                       // causal cumsum of k.*v
      float a = sq * ctx[i];                   // q .* context  (in {0..4})

      int row = m0 + lhalf * 8 + i;
      int col = n0 + lrow;
      arow_base[(size_t)row * DIM + col] = (_Float16)a;
    }
  }
}

// --------------------------- output projection -----------------------------
// out = a @ Wp.T + bp over (TOTROWS x DIM) rows, fp32 result.
__global__ __launch_bounds__(256) void out_proj_kernel(
    const _Float16* __restrict__ ah,   // (TOTROWS, DIM) f16
    const _Float16* __restrict__ wp,   // (DIM, DIM) f16
    const float* __restrict__ bp,      // (DIM,) fp32
    float* __restrict__ out) {         // (TOTROWS, DIM) fp32
  const int lane  = threadIdx.x & 31;
  const int wave  = threadIdx.x >> 5;
  const int wm    = wave & 3;
  const int wn    = wave >> 2;
  const int m0    = blockIdx.x * 64 + wm * 16;
  const int n0    = blockIdx.y * 32 + wn * 16;
  const int lrow  = lane & 15;
  const int lhalf = lane >> 4;

  v8f acc = (v8f){0.f, 0.f, 0.f, 0.f, 0.f, 0.f, 0.f, 0.f};
  const int arow = m0 + lrow;
  const int brow = n0 + lrow;

  for (int k0 = 0; k0 < DIM; k0 += BK) {
    v16h af = load_frag_a(ah, arow, k0, lhalf);
    v16h bf = load_frag_b(wp, brow, k0, lhalf);
    acc = WMMA_F16(af, bf, acc);
  }

  const int col = n0 + lrow;
  const float bias = bp[col];
#pragma unroll
  for (int i = 0; i < 8; ++i) {
    int row = m0 + lhalf * 8 + i;
    out[(size_t)row * DIM + col] = acc[i] + bias;
  }
}

// ------------------------------- launcher ----------------------------------
extern "C" void kernel_launch(void* const* d_in, const int* in_sizes, int n_in,
                              void* d_out, int out_size, void* d_ws, size_t ws_size,
                              hipStream_t stream) {
  const float* x  = (const float*)d_in[0];   // (T,B,N,D)
  const float* Wq = (const float*)d_in[1];
  const float* Wk = (const float*)d_in[2];
  const float* Wv = (const float*)d_in[3];
  const float* Wp = (const float*)d_in[4];
  const float* bp = (const float*)d_in[5];
  float* out = (float*)d_out;

  // Workspace layout (f16 elements):
  //   xh  : TOTROWS*DIM          (64 MB)
  //   wqh/wkh/wvh/wph : DIM*DIM  (0.5 MB each)
  //   ah  : TOTROWS*DIM          (64 MB)
  const size_t XN = (size_t)TOTROWS * DIM;   // 33,554,432
  const size_t WN = (size_t)DIM * DIM;       // 262,144
  _Float16* xh  = (_Float16*)d_ws;
  _Float16* wqh = xh + XN;
  _Float16* wkh = wqh + WN;
  _Float16* wvh = wkh + WN;
  _Float16* wph = wvh + WN;
  _Float16* ah  = wph + WN;

  // fp32 -> f16 conversion (xh resident in L2 afterwards: 64 MB < 192 MB)
  {
    int n8 = (int)(XN / 8);
    cvt_f32_to_f16_kernel<<<(n8 + 255) / 256, 256, 0, stream>>>(x, xh, n8);
    int w8 = (int)(WN / 8);
    int wb = (w8 + 255) / 256;
    cvt_f32_to_f16_kernel<<<wb, 256, 0, stream>>>(Wq, wqh, w8);
    cvt_f32_to_f16_kernel<<<wb, 256, 0, stream>>>(Wk, wkh, w8);
    cvt_f32_to_f16_kernel<<<wb, 256, 0, stream>>>(Wv, wvh, w8);
    cvt_f32_to_f16_kernel<<<wb, 256, 0, stream>>>(Wp, wph, w8);
  }

  // Fused q/k/v projections + LIF + causal linear attention -> ah
  {
    dim3 grid(MROWS / 64, DIM / 32);   // 256 x 16
    qkv_lif_attn_kernel<<<grid, 256, 0, stream>>>(xh, wqh, wkh, wvh, ah);
  }

  // Output projection + bias -> d_out (fp32)
  {
    dim3 grid(TOTROWS / 64, DIM / 32); // 1024 x 16
    out_proj_kernel<<<grid, 256, 0, stream>>>(ah, wph, bp, out);
  }
}
